// DeformConv2d_33285996544418
// MI455X (gfx1250) — matrix operand order repaired and checked by reading, not compiled, not run on hardware
//
#include <hip/hip_runtime.h>

// ---------------------------------------------------------------------------
// Fused modulated deformable conv (DCNv2) for gfx1250 (MI455X).
//   N=8, Cin=256, H=W=64, Cout=256, K=3, stride=1, pad=1, dil=1, groups=1.
// Pass 1: pre-pack weights to bf16 in per-chunk A-tile layout (d_ws, 1.2 MB).
// Pass 2: software-pipelined fused kernel — bilinear sampling + mask in f32,
//         GEMM via v_wmma_f32_16x16x32_bf16, double-buffered LDS tiles so
//         next-chunk gathers overlap the WMMAs. One 256-thread WG (8 wave32)
//         per (image, output-row): 64 pixels x 256 Cout, K = 2304 in 72
//         chunks of 32 (tap-major, channel-minor).
// ---------------------------------------------------------------------------

typedef __attribute__((ext_vector_type(16))) __bf16 v16bf;
typedef __attribute__((ext_vector_type(8)))  __bf16 v8bf;
typedef __attribute__((ext_vector_type(2)))  __bf16 v2bf;
typedef __attribute__((ext_vector_type(8)))  float  v8f;

#define CIN       256
#define COUT      256
#define HH        64
#define WW        64
#define TAPS      9
#define NPIX      64                       // output pixels per WG (one row)
#define KTILE     32
#define NTHREADS  256
#define NCHUNKS   ((CIN * TAPS) / KTILE)   // 72
#define ATILE     (COUT * KTILE)           // 8192 bf16 = 16 KB per chunk

// ---- pass 1: weight[o][c][k] f32  ->  Wp[chunk][o][cl] bf16 ----------------
__global__ void dcn_pack_weights(const float* __restrict__ weight,
                                 __bf16* __restrict__ wpacked)
{
    const int e     = blockIdx.x * NTHREADS + threadIdx.x; // 0 .. 589823
    const int chunk = e >> 13;            // 8192 elements per chunk
    const int r     = e & 8191;
    const int o     = r >> 5;
    const int cl    = r & 31;
    const int k     = chunk >> 3;
    const int c     = ((chunk & 7) << 5) + cl;
    wpacked[e] = (__bf16)weight[((o << 8) + c) * TAPS + k];
}

// ---- pass 2: fused sample + WMMA GEMM, software pipelined ------------------
__launch_bounds__(NTHREADS, 1)
__global__ void dcn_fused_wmma_bf16(const float* __restrict__ input,
                                    const float* __restrict__ offset,
                                    const float* __restrict__ mask,
                                    const __bf16* __restrict__ wpacked,
                                    const float* __restrict__ bias,
                                    float* __restrict__ out)
{
    // Double-buffered tiles.
    __shared__ __align__(16) __bf16 aLds[2][ATILE];         // 32 KB
    __shared__ __align__(16) __bf16 bLds[2][NPIX * KTILE];  //  8 KB
    // Bilinear descriptors per (pixel, tap): 4 clamped in-plane element
    // indices + 4 premultiplied (bilinear * mask, zeroed-if-OOB) weights.
    __shared__ int   dAddr[NPIX * TAPS][4];                 //  9 KB
    __shared__ float dWt  [NPIX * TAPS][4];                 //  9 KB

    const int tid     = threadIdx.x;
    const int lane    = tid & 31;          // wave32
    const int wave    = tid >> 5;          // 0..7
    const int pixBase = blockIdx.x * NPIX; // one output row (Wo == 64)
    const int n       = blockIdx.y;        // image

    // ---- stage 0: build bilinear descriptors -----------------------------
    for (int d = tid; d < NPIX * TAPS; d += NTHREADS) {
        const int point = d / TAPS;
        const int k     = d - point * TAPS;
        const int pix   = pixBase + point;
        const int h     = pix >> 6;        // Wo == 64
        const int w     = pix & 63;
        const int ky    = k / 3;
        const int kx    = k - ky * 3;

        const float offy = offset[((n * 18 + 2 * k    ) << 12) + pix];
        const float offx = offset[((n * 18 + 2 * k + 1) << 12) + pix];
        const float m    = mask  [((n *  9 +     k    ) << 12) + pix];

        const float sy  = (float)(h + ky - 1) + offy;   // stride 1, pad 1
        const float sx  = (float)(w + kx - 1) + offx;
        const float y0f = floorf(sy), x0f = floorf(sx);
        const float wy1 = sy - y0f,  wx1 = sx - x0f;
        const float wy0 = 1.0f - wy1, wx0 = 1.0f - wx1;
        const int   y0  = (int)y0f,  x0  = (int)x0f;

#pragma unroll
        for (int j = 0; j < 4; ++j) {
            const int   dy = j >> 1, dx = j & 1;
            const int   yi = y0 + dy, xi = x0 + dx;
            float wv = (dy ? wy1 : wy0) * (dx ? wx1 : wx0) * m;
            if (yi < 0 || yi >= HH || xi < 0 || xi >= WW) wv = 0.0f;
            const int yc = min(max(yi, 0), HH - 1);
            const int xc = min(max(xi, 0), WW - 1);
            dAddr[d][j] = yc * WW + xc;    // in-plane element index
            dWt [d][j] = wv;               // mask folded in; 0 if OOB
        }
    }

    // ---- accumulators: wave owns Cout rows [wave*32, wave*32+32) ---------
    v8f acc[2][4];
#pragma unroll
    for (int mi = 0; mi < 2; ++mi)
#pragma unroll
        for (int ni = 0; ni < 4; ++ni)
            acc[mi][ni] = (v8f)(0.0f);

    const int bC2     = (tid & 15) << 1;   // channel pair within chunk
    const int bPoint  = tid >> 4;          // pixel within pass group
    const int chanOff = bC2 << 12;         // per-thread element offset
    const int half    = lane >> 4;         // WMMA lane half
    const int r       = lane & 15;

    const v8bf* __restrict__ wpv = (const v8bf*)wpacked;

    // Pipeline registers: staged A tile slice + staged gathers/weights.
    v8bf  aw[4];
    float g[4][8];     // [pass][plane0 c0..c3 | plane1 c0..c3]
    float wreg[4][4];  // [pass][corner]

    __syncthreads();   // descriptors ready

    // ---- prologue: issue chunk 0 loads -----------------------------------
    {
        const float* __restrict__ base =
            input + ((size_t)(n << 8) << 12);          // chunk 0: cBase = 0
#pragma unroll
        for (int q = 0; q < 4; ++q)
            aw[q] = wpv[q * NTHREADS + tid];
#pragma unroll
        for (int p = 0; p < 4; ++p) {
            const int point = p * 16 + bPoint;
            const int d     = point * TAPS;            // chunk 0: k = 0
#pragma unroll
            for (int j = 0; j < 4; ++j) {
                const int idx = chanOff + dAddr[d][j]; // 32-bit scaled index
                wreg[p][j]  = dWt[d][j];
                g[p][j]     = base[idx];
                g[p][4 + j] = base[idx + (HH * WW)];
            }
        }
    }

    // ---- main pipelined K loop -------------------------------------------
    for (int chunk = 0; chunk < NCHUNKS; ++chunk) {
        const int cur = chunk & 1;
        __bf16* __restrict__ aBuf = aLds[cur];
        __bf16* __restrict__ bBuf = bLds[cur];

        // Drain staged data for this chunk into LDS (loadcnt waits land here).
        {
            v8bf* __restrict__ al = (v8bf*)aBuf;
#pragma unroll
            for (int q = 0; q < 4; ++q)
                al[q * NTHREADS + tid] = aw[q];
#pragma unroll
            for (int p = 0; p < 4; ++p) {
                const int point = p * 16 + bPoint;
                const float v0 = g[p][0] * wreg[p][0] + g[p][1] * wreg[p][1]
                               + g[p][2] * wreg[p][2] + g[p][3] * wreg[p][3];
                const float v1 = g[p][4] * wreg[p][0] + g[p][5] * wreg[p][1]
                               + g[p][6] * wreg[p][2] + g[p][7] * wreg[p][3];
                *(v2bf*)&bBuf[(point << 5) + bC2] =
                    (v2bf){(__bf16)v0, (__bf16)v1};
            }
        }
        __syncthreads();

        // Issue next chunk's loads; results consumed next iteration, so the
        // WMMAs below overlap the global-memory latency.
        if (chunk + 1 < NCHUNKS) {
            const int nk  = (chunk + 1) >> 3;
            const int ncb = ((chunk + 1) & 7) << 5;
            const float* __restrict__ base =
                input + ((size_t)((n << 8) + ncb) << 12);  // wave-uniform
#pragma unroll
            for (int q = 0; q < 4; ++q)
                aw[q] = wpv[(size_t)(chunk + 1) * (ATILE / 8)
                            + q * NTHREADS + tid];
#pragma unroll
            for (int p = 0; p < 4; ++p) {
                const int point = p * 16 + bPoint;
                const int d     = point * TAPS + nk;
#pragma unroll
                for (int j = 0; j < 4; ++j) {
                    const int idx = chanOff + dAddr[d][j];
                    wreg[p][j]  = dWt[d][j];
                    g[p][j]     = base[idx];
                    g[p][4 + j] = base[idx + (HH * WW)];
                }
            }
        }

        // B fragments (contiguous k order used here: lane<16 -> K0..15, lane>=16 ->
        // K16..31, K-contiguous elements). bBuf is K-contiguous per pixel.
        v16bf bfrag[4];
#pragma unroll
        for (int ni = 0; ni < 4; ++ni) {
            const __bf16* bp = &bBuf[(((ni << 4) + r) << 5) + (half << 4)];
            const v8bf lo = *(const v8bf*)(bp);
            const v8bf hi = *(const v8bf*)(bp + 8);
            bfrag[ni] = __builtin_shufflevector(lo, hi,
                0, 1, 2, 3, 4, 5, 6, 7, 8, 9, 10, 11, 12, 13, 14, 15);
        }

        // A fragments + WMMA (A layout: lane<16 -> K0..15,
        // lane>=16 -> K16..31, matching B)
#pragma unroll
        for (int mi = 0; mi < 2; ++mi) {
            const int row = (wave << 5) + (mi << 4) + r;
            const __bf16* ap = &aBuf[(row << 5) + (half << 4)];
            const v8bf lo = *(const v8bf*)(ap);
            const v8bf hi = *(const v8bf*)(ap + 8);
            const v16bf afrag = __builtin_shufflevector(lo, hi,
                0, 1, 2, 3, 4, 5, 6, 7, 8, 9, 10, 11, 12, 13, 14, 15);
#pragma unroll
            for (int ni = 0; ni < 4; ++ni) {
                acc[mi][ni] = __builtin_amdgcn_wmma_f32_16x16x32_bf16(
                    false, afrag, false, bfrag[ni],
                    (short)0, acc[mi][ni], false, false);
            }
        }
    }

    // ---- epilogue: bias + store (C layout: VGPR r -> M=r / M=8+r) --------
    const int colN   = lane & 15;
    const int rowOff = (lane >> 4) * 8;
#pragma unroll
    for (int mi = 0; mi < 2; ++mi) {
#pragma unroll
        for (int ni = 0; ni < 4; ++ni) {
            union { v8f v; float f[8]; } u;
            u.v = acc[mi][ni];
            const int oBase = (wave << 5) + (mi << 4) + rowOff;
            const int pix   = pixBase + (ni << 4) + colN;
#pragma unroll
            for (int rr = 0; rr < 8; ++rr) {
                const int o = oBase + rr;
                out[(((n << 8) + o) << 12) + pix] = u.f[rr] + bias[o];
            }
        }
    }
}

extern "C" void kernel_launch(void* const* d_in, const int* in_sizes, int n_in,
                              void* d_out, int out_size, void* d_ws, size_t ws_size,
                              hipStream_t stream) {
    (void)in_sizes; (void)n_in; (void)ws_size; (void)out_size;
    const float* input  = (const float*)d_in[0];
    const float* offset = (const float*)d_in[1];
    const float* mask   = (const float*)d_in[2];
    const float* weight = (const float*)d_in[3];
    const float* bias   = (const float*)d_in[4];
    float*  out     = (float*)d_out;
    __bf16* wpacked = (__bf16*)d_ws;    // 2304*256*2 B = 1.18 MB scratch

    // Pass 1: pack weights (same stream -> ordered before the fused pass).
    dcn_pack_weights<<<dim3(NCHUNKS * ATILE / NTHREADS), NTHREADS, 0, stream>>>(
        weight, wpacked);

    // Pass 2: fused deformable conv.
    dim3 grid(64, 8);   // (Ho*Wo / 64 pixel rows, N images)
    dcn_fused_wmma_bf16<<<grid, NTHREADS, 0, stream>>>(
        input, offset, mask, wpacked, bias, out);
}
